// GraphormerLayer_53163105190239
// MI455X (gfx1250) — matrix operand order repaired and checked by reading, not compiled, not run on hardware
//
#include <hip/hip_runtime.h>
#include <hip/hip_bf16.h>
#include <cstdint>
#include <cstddef>

#ifndef __has_builtin
#define __has_builtin(x) 0
#endif
#if __has_builtin(__builtin_amdgcn_tensor_load_to_lds)
#define HAVE_TDM 1
#else
#define HAVE_TDM 0
#endif
#if defined(__has_include)
#if __has_include(<hip/amd_detail/amd_gfx1250_TDM.h>)
#define TDM_6ARG 1
#endif
#endif
#ifndef TDM_6ARG
#define TDM_6ARG 0
#endif

// ---------------------------------------------------------------------------
// Types for CDNA5 WMMA (wave32): v16bf A/B fragments, v8f C/D accumulators
// ---------------------------------------------------------------------------
typedef __bf16 bf16_t;
typedef bf16_t v16bf __attribute__((ext_vector_type(16)));
typedef float  v8f   __attribute__((ext_vector_type(8)));
typedef unsigned int u32x4 __attribute__((ext_vector_type(4)));
typedef unsigned int uint32x4_t __attribute__((ext_vector_type(4)));
typedef int int32x8_t __attribute__((ext_vector_type(8)));
typedef int int32x4_t __attribute__((ext_vector_type(4)));

union FragB16 { v16bf v; u32x4 u[2]; };

__device__ inline bf16_t f2bf(float f) {
  unsigned u = __builtin_bit_cast(unsigned, f);
  u += 0x7FFFu + ((u >> 16) & 1u);            // round-to-nearest-even
  unsigned short h = (unsigned short)(u >> 16);
  return __builtin_bit_cast(bf16_t, h);
}

// A-matrix 16x32 bf16 fragment (row-major source, row pointer given):
// lane<16: K = [k0+0..7],[k0+16..23]; lane>=16: K = [k0+8..15],[k0+24..31]
__device__ inline v16bf load_a_frag(const bf16_t* row, int k0, int half) {
  FragB16 f;
  f.u[0] = *(const u32x4*)(row + k0 + 8 * half);
  f.u[1] = *(const u32x4*)(row + k0 + 8 * half + 16);
  return f.v;
}

// B-matrix 32x16 bf16 fragment from B^T row-major (row = column of B):
// lane<16: K = [k0+0..7],[k0+16..23]; lane>=16: K = [k0+8..15],[k0+24..31]
__device__ inline v16bf load_b_frag(const bf16_t* row, int k0, int half) {
  FragB16 f;
  f.u[0] = *(const u32x4*)(row + k0 + 8 * half);
  f.u[1] = *(const u32x4*)(row + k0 + 8 * half + 16);
  return f.v;
}

__device__ inline v8f wmma_bf16(v16bf a, v16bf b, v8f c) {
  return __builtin_amdgcn_wmma_f32_16x16x32_bf16(
      false, a, false, b, (short)0, c, false, false);
}

// ---------------------------------------------------------------------------
// Tensor Data Mover: async 2D tile (bf16) global -> LDS, rows padded so the
// LDS row stride is 20 DWORDs (40 bf16) -> conflict-free B128 fragment reads.
// D# layout per CDNA5 ISA ch.8 (group0: count/lds/global/type, group1: dims).
// ---------------------------------------------------------------------------
#define LDS_STRIDE 40   // 32 data + 8 pad bf16 per row (16 DW + 4 DW pad)

#if HAVE_TDM
__device__ inline void tdm_load_2d(const void* gptr, unsigned lds_off,
                                   unsigned td0, unsigned td1,
                                   unsigned tile0, unsigned tile1,
                                   unsigned long long stride0) {
  unsigned long long ga = (unsigned long long)(uintptr_t)gptr;
  uint32x4_t g0 = { 1u,                                   // count=1, user mode
                    lds_off,                              // LDS byte address
                    (unsigned)(ga & 0xffffffffull),
                    (unsigned)((ga >> 32) & 0x1ffffffull) | (2u << 30) }; // type=2
  int w0 = (1 << 16)           // data_size = 2 bytes
         | (1 << 20)           // pad_enable
         | (3 << 22)           // pad_interval: 16 DWORDs (one 32-elem bf16 row)
         | (3 << 25);          // pad_amount: 4 DWORDs
  int w1 = (int)((td0 & 0xffffu) << 16);                          // tensor_dim0 lo
  int w2 = (int)(((td0 >> 16) & 0xffffu) | ((td1 & 0xffffu) << 16));
  int w3 = (int)(((td1 >> 16) & 0xffffu) | ((tile0 & 0xffffu) << 16));
  int w4 = (int)(tile1 & 0xffffu);                                // tile_dim2 = 0
  int w5 = (int)(stride0 & 0xffffffffull);
  int w6 = (int)((stride0 >> 32) & 0xffffull);                    // stride1 = 0
  int32x8_t g1 = { w0, w1, w2, w3, w4, w5, w6, 0 };
  int32x4_t z4 = { 0, 0, 0, 0 };
#if TDM_6ARG
  int32x8_t z8 = { 0, 0, 0, 0, 0, 0, 0, 0 };
  __builtin_amdgcn_tensor_load_to_lds(g0, g1, z4, z4, z8, 0);
#else
  __builtin_amdgcn_tensor_load_to_lds(g0, g1, z4, z4, 0);
#endif
}
#endif

// ---------------------------------------------------------------------------
// Weight convert + transpose: W[K,N] fp32 -> WT[N,K] bf16
// ---------------------------------------------------------------------------
__global__ void tw_conv_kernel(const float* __restrict__ W, bf16_t* __restrict__ WT,
                               int K, int N) {
  int idx = blockIdx.x * 256 + threadIdx.x;
  if (idx >= N * K) return;
  int n = idx / K, k = idx % K;
  WT[idx] = f2bf(W[(size_t)k * N + n]);
}

// ---------------------------------------------------------------------------
// LayerNorm over D=512, one wave per row, output bf16
// ---------------------------------------------------------------------------
__global__ __launch_bounds__(256)
void ln_bf16_kernel(const float* __restrict__ x, const float* __restrict__ w,
                    const float* __restrict__ b, bf16_t* __restrict__ out, int rows) {
  int lane = threadIdx.x & 31;
  int row  = blockIdx.x * 8 + (threadIdx.x >> 5);
  if (row >= rows) return;
  const float* xr = x + (size_t)row * 512;
  float vals[16], s = 0.f, s2 = 0.f;
#pragma unroll
  for (int i = 0; i < 16; ++i) {
    float t = xr[lane + i * 32];
    vals[i] = t; s += t; s2 += t * t;
  }
#pragma unroll
  for (int m = 16; m; m >>= 1) {
    s  += __shfl_xor(s,  m, 32);
    s2 += __shfl_xor(s2, m, 32);
  }
  float mu  = s * (1.f / 512.f);
  float var = s2 * (1.f / 512.f) - mu * mu;
  float rs  = rsqrtf(var + 1e-5f);
  bf16_t* orow = out + (size_t)row * 512;
#pragma unroll
  for (int i = 0; i < 16; ++i) {
    int c = lane + i * 32;
    orow[c] = f2bf((vals[i] - mu) * rs * w[c] + b[c]);
  }
}

// ---------------------------------------------------------------------------
// bf16 WMMA GEMM, LDS-tiled + double-buffered, staged with async TDM loads.
// C[M,N] = A[M,K] @ BT[N,K]^T + bias
// mode 0: out bf16;  mode 1: out bf16 + ReLU;  mode 2: out f32 + residual
// Block: 256 thr (8 waves). Block tile 128(M) x 64(N); wave tile 16 x 64.
// ---------------------------------------------------------------------------
__global__ __launch_bounds__(256)
void gemm_bf16_kernel(const bf16_t* __restrict__ A, const bf16_t* __restrict__ BT,
                      const float* __restrict__ bias, const float* __restrict__ resid,
                      float* __restrict__ outF, bf16_t* __restrict__ outB,
                      int M, int N, int K, int mode) {
  __shared__ bf16_t Asl[2][128][LDS_STRIDE];
  __shared__ bf16_t Bsl[2][64][LDS_STRIDE];
  int lane = threadIdx.x & 31;
  int wv   = threadIdx.x >> 5;
  int tilesN = N >> 6;
  int tm = blockIdx.x / tilesN, tn = blockIdx.x % tilesN;
  int colN = lane & 15, half = lane >> 4;
  const bf16_t* Ablk = A + (size_t)tm * 128 * K;
  const bf16_t* Bblk = BT + (size_t)tn * 64 * K;

  v8f acc[4] = {};
  int nsteps = K >> 5;

  // ---- stage helper (TDM async when available, cooperative copy otherwise)
  auto stage = [&](int buf, int k0) {
#if HAVE_TDM
    if (threadIdx.x < 32) {
      tdm_load_2d(Ablk + k0, (unsigned)(uintptr_t)&Asl[buf][0][0],
                  (unsigned)(K - k0), 128u, 32u, 128u, (unsigned long long)K);
      tdm_load_2d(Bblk + k0, (unsigned)(uintptr_t)&Bsl[buf][0][0],
                  (unsigned)(K - k0), 64u, 32u, 64u, (unsigned long long)K);
    }
#else
    for (int c = threadIdx.x; c < 512; c += 256) {     // A: 128 rows x 4 chunks
      int r = c >> 2, cc = c & 3;
      *(u32x4*)&Asl[buf][r][cc * 8] =
          *(const u32x4*)(Ablk + (size_t)r * K + k0 + cc * 8);
    }
    {                                                   // B: 64 rows x 4 chunks
      int r = threadIdx.x >> 2, cc = threadIdx.x & 3;
      *(u32x4*)&Bsl[buf][r][cc * 8] =
          *(const u32x4*)(Bblk + (size_t)r * K + k0 + cc * 8);
    }
#endif
  };
  auto finish = [&]() {
#if HAVE_TDM
    if (threadIdx.x < 32) __builtin_amdgcn_s_wait_tensorcnt(0);
#endif
    __syncthreads();
  };

  stage(0, 0);
  finish();

  for (int s = 0; s < nsteps; ++s) {
    int buf = s & 1;
    if (s + 1 < nsteps) stage(buf ^ 1, (s + 1) * 32);   // async prefetch
    // compute on current buffer
    v16bf af = load_a_frag(&Asl[buf][wv * 16 + colN][0], 0, half);
    v16bf bfv[4];
#pragma unroll
    for (int nt = 0; nt < 4; ++nt)
      bfv[nt] = load_b_frag(&Bsl[buf][nt * 16 + colN][0], 0, half);
#pragma unroll
    for (int nt = 0; nt < 4; ++nt)
      acc[nt] = wmma_bf16(af, bfv[nt], acc[nt]);
    finish();                                           // wait DMA + barrier
  }

#pragma unroll
  for (int nt = 0; nt < 4; ++nt) {
    int c = tn * 64 + nt * 16 + colN;
    float bv = bias[c];
#pragma unroll
    for (int v = 0; v < 8; ++v) {
      int r = tm * 128 + wv * 16 + v + 8 * half;        // C layout: row = v+8*half
      float val = acc[nt][v] + bv;
      if (mode == 1) val = fmaxf(val, 0.f);
      size_t idx = (size_t)r * N + c;
      if (mode == 2) outF[idx] = val + resid[idx];
      else           outB[idx] = f2bf(val);
    }
  }
}

// ---------------------------------------------------------------------------
// Transpose V out of qkv buffer: vT[b][h][d][n] (bf16, [64][1024] per head)
// ---------------------------------------------------------------------------
__global__ void vtrans_kernel(const bf16_t* __restrict__ qkv, bf16_t* __restrict__ vT) {
  int idx = blockIdx.x * 256 + threadIdx.x;       // 2^22 elements
  int n = idx & 1023;
  int d = (idx >> 10) & 63;
  int h = (idx >> 16) & 7;
  int b = idx >> 19;
  vT[idx] = qkv[((size_t)(b * 1024 + n)) * 1536 + 1024 + h * 64 + d];
}

// ---------------------------------------------------------------------------
// Flash attention: one wave = 16 queries of one (b,h); 64 keys per step.
// qkv: bf16 [B*N,1536] (Q|K|V); vT: bf16 [B*H][64][1024]; bias: f32 [B*H][N][N]
// out: bf16 [B*N,512]
// ---------------------------------------------------------------------------
__global__ __launch_bounds__(256)
void attn_kernel(const bf16_t* __restrict__ qkv, const bf16_t* __restrict__ vT,
                 const float* __restrict__ bias, bf16_t* __restrict__ outA) {
  __shared__ bf16_t pbuf[8][16][72];              // per-wave P tile, padded rows
  int lane = threadIdx.x & 31;
  int wave = threadIdx.x >> 5;
  int colN = lane & 15, half = lane >> 4;
  int bh   = blockIdx.x >> 3;                     // 0..63
  int qblk = blockIdx.x & 7;
  int b = bh >> 3, h = bh & 7;
  int q0 = qblk * 128 + wave * 16;

  const bf16_t* qbase = qkv + ((size_t)(b * 1024 + q0 + colN)) * 1536 + h * 64;
  const bf16_t* kbase = qkv + ((size_t)b * 1024) * 1536 + 512 + h * 64;
  const bf16_t* vtb   = vT + (size_t)bh * 64 * 1024;
  const float*  bsb   = bias + (size_t)bh * 1024 * 1024;

  v16bf qf0 = load_a_frag(qbase, 0, half);
  v16bf qf1 = load_a_frag(qbase, 32, half);

  v8f O[4] = {};
  float m[8], l[8];
#pragma unroll
  for (int v = 0; v < 8; ++v) { m[v] = -3.0e38f; l[v] = 0.f; }

  for (int j = 0; j < 1024; j += 64) {
    // batch all K-fragment loads first (one clause, one wait)
    v16bf kf[4][2];
#pragma unroll
    for (int t = 0; t < 4; ++t) {
      const bf16_t* krow = kbase + (size_t)(j + t * 16 + colN) * 1536;
      kf[t][0] = load_b_frag(krow, 0, half);
      kf[t][1] = load_b_frag(krow, 32, half);
    }
    v8f S[4];
#pragma unroll
    for (int t = 0; t < 4; ++t) {
      v8f s = {};
      s = wmma_bf16(qf0, kf[t][0], s);
      s = wmma_bf16(qf1, kf[t][1], s);
      S[t] = s;
    }
    // scale + bias (this is the machine's dominant HBM stream)
#pragma unroll
    for (int t = 0; t < 4; ++t)
#pragma unroll
      for (int v = 0; v < 8; ++v) {
        int r = q0 + v + 8 * half;
        float bv = bsb[(size_t)r * 1024 + j + t * 16 + colN];
        S[t][v] = S[t][v] * 0.125f + bv;
      }
    // online softmax per row (row axis: v + 8*half; key axis: lanes in half)
    float ef[8];
#pragma unroll
    for (int v = 0; v < 8; ++v) {
      float rm = fmaxf(fmaxf(S[0][v], S[1][v]), fmaxf(S[2][v], S[3][v]));
#pragma unroll
      for (int msk = 8; msk; msk >>= 1) rm = fmaxf(rm, __shfl_xor(rm, msk, 16));
      float mn = fmaxf(m[v], rm);
      float e  = __expf(m[v] - mn);
      m[v] = mn; ef[v] = e;
      float ps = 0.f;
#pragma unroll
      for (int t = 0; t < 4; ++t) {
        float p = __expf(S[t][v] - mn);
        S[t][v] = p; ps += p;
      }
#pragma unroll
      for (int msk = 8; msk; msk >>= 1) ps += __shfl_xor(ps, msk, 16);
      l[v] = l[v] * e + ps;
    }
#pragma unroll
    for (int d = 0; d < 4; ++d)
#pragma unroll
      for (int v = 0; v < 8; ++v) O[d][v] *= ef[v];
    // P (C layout) -> LDS row-major -> A fragments
#pragma unroll
    for (int t = 0; t < 4; ++t)
#pragma unroll
      for (int v = 0; v < 8; ++v)
        pbuf[wave][v + 8 * half][t * 16 + colN] = f2bf(S[t][v]);
#pragma unroll
    for (int kk = 0; kk < 2; ++kk) {
      v16bf pf = load_a_frag(&pbuf[wave][colN][0], kk * 32, half);
      v16bf vf[4];
#pragma unroll
      for (int d = 0; d < 4; ++d) {
        const bf16_t* vrow = vtb + (size_t)(d * 16 + colN) * 1024 + j + kk * 32;
        vf[d] = load_b_frag(vrow, 0, half);
      }
#pragma unroll
      for (int d = 0; d < 4; ++d)
        O[d] = wmma_bf16(pf, vf[d], O[d]);
    }
  }
  // normalize + store bf16 [B*N, 512]
#pragma unroll
  for (int v = 0; v < 8; ++v) {
    float inv = 1.f / l[v];
    int r = q0 + v + 8 * half;
    bf16_t* orow = outA + ((size_t)(b * 1024 + r)) * 512 + h * 64;
#pragma unroll
    for (int d = 0; d < 4; ++d)
      orow[d * 16 + colN] = f2bf(O[d][v] * inv);
  }
}

// ---------------------------------------------------------------------------
// Host launch
// ---------------------------------------------------------------------------
extern "C" void kernel_launch(void* const* d_in, const int* in_sizes, int n_in,
                              void* d_out, int out_size, void* d_ws, size_t ws_size,
                              hipStream_t stream) {
  (void)in_sizes; (void)n_in; (void)out_size; (void)ws_size;
  const float* x      = (const float*)d_in[0];
  const float* abias  = (const float*)d_in[1];
  const float* qkv_w  = (const float*)d_in[2];
  const float* qkv_b  = (const float*)d_in[3];
  const float* proj_w = (const float*)d_in[4];
  const float* proj_b = (const float*)d_in[5];
  const float* ffn1_w = (const float*)d_in[6];
  const float* ffn1_b = (const float*)d_in[7];
  const float* ffn2_w = (const float*)d_in[8];
  const float* ffn2_b = (const float*)d_in[9];
  const float* ln1_w  = (const float*)d_in[10];
  const float* ln1_b  = (const float*)d_in[11];
  const float* ln2_w  = (const float*)d_in[12];
  const float* ln2_b  = (const float*)d_in[13];
  float* out = (float*)d_out;

  char* ws = (char*)d_ws;
  size_t off = 0;
  auto alloc = [&](size_t bytes) -> char* {
    char* p = ws + off;
    off += (bytes + 255) & ~(size_t)255;
    return p;
  };
  bf16_t* h1      = (bf16_t*)alloc(8192ull * 512 * 2);
  bf16_t* qkvb    = (bf16_t*)alloc(8192ull * 1536 * 2);
  bf16_t* vT      = (bf16_t*)alloc(64ull * 64 * 1024 * 2);
  bf16_t* aout    = (bf16_t*)alloc(8192ull * 512 * 2);
  float*  x2      = (float*) alloc(8192ull * 512 * 4);
  bf16_t* h2      = (bf16_t*)alloc(8192ull * 512 * 2);
  bf16_t* mid     = (bf16_t*)alloc(8192ull * 2048 * 2);
  bf16_t* wt_qkv  = (bf16_t*)alloc(1536ull * 512 * 2);
  bf16_t* wt_proj = (bf16_t*)alloc(512ull * 512 * 2);
  bf16_t* wt_f1   = (bf16_t*)alloc(2048ull * 512 * 2);
  bf16_t* wt_f2   = (bf16_t*)alloc(512ull * 2048 * 2);

  // 1) weights -> bf16, transposed [N,K]
  tw_conv_kernel<<<(1536 * 512 + 255) / 256, 256, 0, stream>>>(qkv_w, wt_qkv, 512, 1536);
  tw_conv_kernel<<<(512 * 512 + 255) / 256, 256, 0, stream>>>(proj_w, wt_proj, 512, 512);
  tw_conv_kernel<<<(2048 * 512 + 255) / 256, 256, 0, stream>>>(ffn1_w, wt_f1, 512, 2048);
  tw_conv_kernel<<<(512 * 2048 + 255) / 256, 256, 0, stream>>>(ffn2_w, wt_f2, 2048, 512);

  // 2) LN1
  ln_bf16_kernel<<<1024, 256, 0, stream>>>(x, ln1_w, ln1_b, h1, 8192);

  // 3) QKV GEMM: [8192,512] @ [512,1536]   grid = (M/128)*(N/64)
  gemm_bf16_kernel<<<64 * 24, 256, 0, stream>>>(h1, wt_qkv, qkv_b, nullptr,
                                                nullptr, qkvb, 8192, 1536, 512, 0);

  // 4) V transpose per head
  vtrans_kernel<<<(1 << 22) / 256, 256, 0, stream>>>(qkvb, vT);

  // 5) flash attention
  attn_kernel<<<512, 256, 0, stream>>>(qkvb, vT, abias, aout);

  // 6) proj + residual -> x2 (f32)
  gemm_bf16_kernel<<<64 * 8, 256, 0, stream>>>(aout, wt_proj, proj_b, x,
                                               x2, nullptr, 8192, 512, 512, 2);

  // 7) LN2
  ln_bf16_kernel<<<1024, 256, 0, stream>>>(x2, ln2_w, ln2_b, h2, 8192);

  // 8) FFN1 + ReLU
  gemm_bf16_kernel<<<64 * 32, 256, 0, stream>>>(h2, wt_f1, ffn1_b, nullptr,
                                                nullptr, mid, 8192, 2048, 512, 1);

  // 9) FFN2 + residual -> out (f32)
  gemm_bf16_kernel<<<64 * 8, 256, 0, stream>>>(mid, wt_f2, ffn2_b, x2,
                                               out, nullptr, 8192, 512, 2048, 2);
}
